// PVConv_27711128994385
// MI455X (gfx1250) — matrix operand order repaired and checked by reading, not compiled, not run on hardware
//
#include <hip/hip_runtime.h>

#define B_  4
#define N_  16384
#define CIN_ 64
#define COUT_ 128
#define R_  32
#define NVOX (B_ * R_ * R_ * R_)          // 131072

typedef __attribute__((ext_vector_type(16))) __bf16 vbf16;
typedef __attribute__((ext_vector_type(8)))  float  v8f;

// ---------- helpers ----------
__device__ __forceinline__ unsigned short f2bf(float f) {
    unsigned int u = __float_as_uint(f);
    u += 0x7FFFu + ((u >> 16) & 1u);      // round-to-nearest-even
    return (unsigned short)(u >> 16);
}
__device__ __forceinline__ float bf2f(unsigned short h) {
    return __uint_as_float(((unsigned int)h) << 16);
}

// ---------- 0: zero counts+sums region ----------
__global__ void k_zero(uint4* __restrict__ p, size_t n4) {
    size_t i = blockIdx.x * (size_t)blockDim.x + threadIdx.x;
    size_t stride = gridDim.x * (size_t)blockDim.x;
    uint4 z; z.x = z.y = z.z = z.w = 0u;
    for (; i < n4; i += stride) p[i] = z;
}

// ---------- 1: per-batch mean + max-norm ----------
__global__ void k_stats(const float* __restrict__ coords, float* __restrict__ stats) {
    int b = blockIdx.x, tid = threadIdx.x;
    __shared__ float sx[256], sy[256], sz[256];
    __shared__ float mean[3];
    const float* cb = coords + (size_t)b * N_ * 3;
    float ax = 0.f, ay = 0.f, az = 0.f;
    for (int i = tid; i < N_; i += 256) {
        ax += cb[i * 3 + 0]; ay += cb[i * 3 + 1]; az += cb[i * 3 + 2];
    }
    sx[tid] = ax; sy[tid] = ay; sz[tid] = az; __syncthreads();
    for (int off = 128; off > 0; off >>= 1) {
        if (tid < off) { sx[tid] += sx[tid + off]; sy[tid] += sy[tid + off]; sz[tid] += sz[tid + off]; }
        __syncthreads();
    }
    if (tid == 0) {
        mean[0] = sx[0] * (1.f / N_); mean[1] = sy[0] * (1.f / N_); mean[2] = sz[0] * (1.f / N_);
    }
    __syncthreads();
    float mx = 0.f;
    for (int i = tid; i < N_; i += 256) {
        float dx = cb[i * 3 + 0] - mean[0];
        float dy = cb[i * 3 + 1] - mean[1];
        float dz = cb[i * 3 + 2] - mean[2];
        mx = fmaxf(mx, sqrtf(dx * dx + dy * dy + dz * dz));
    }
    sx[tid] = mx; __syncthreads();
    for (int off = 128; off > 0; off >>= 1) {
        if (tid < off) sx[tid] = fmaxf(sx[tid], sx[tid + off]);
        __syncthreads();
    }
    if (tid == 0) {
        stats[b * 4 + 0] = mean[0]; stats[b * 4 + 1] = mean[1]; stats[b * 4 + 2] = mean[2];
        stats[b * 4 + 3] = 2.f * sx[0] + 1e-6f;
    }
}

// ---------- 2: voxelize (scatter add) ----------
__global__ void k_voxelize(const float* __restrict__ coords, const float* __restrict__ feat,
                           const float* __restrict__ stats, float* __restrict__ nc_out,
                           float* __restrict__ sums, float* __restrict__ cnts) {
    unsigned int t = blockIdx.x * blockDim.x + threadIdx.x;
    if (t >= B_ * N_ * 4) return;
    unsigned int sub = t & 3u;
    unsigned int p = t >> 2;
    unsigned int b = p >> 14;
    const float* st = stats + b * 4;
    float inv = 1.f / st[3];
    float n0 = (coords[p * 3 + 0] - st[0]) * inv + 0.5f;
    float n1 = (coords[p * 3 + 1] - st[1]) * inv + 0.5f;
    float n2 = (coords[p * 3 + 2] - st[2]) * inv + 0.5f;
    n0 = fminf(fmaxf(n0 * (float)R_, 0.f), (float)(R_ - 1));
    n1 = fminf(fmaxf(n1 * (float)R_, 0.f), (float)(R_ - 1));
    n2 = fminf(fmaxf(n2 * (float)R_, 0.f), (float)(R_ - 1));
    unsigned int v0 = (unsigned int)(int)rintf(n0);
    unsigned int v1 = (unsigned int)(int)rintf(n1);
    unsigned int v2 = (unsigned int)(int)rintf(n2);
    unsigned int lin = b * (R_ * R_ * R_) + v0 * (R_ * R_) + v1 * R_ + v2;
    if (sub == 0) {
        nc_out[p * 3 + 0] = n0; nc_out[p * 3 + 1] = n1; nc_out[p * 3 + 2] = n2;
        atomicAdd(&cnts[lin], 1.0f);
    }
    const float* f = feat + (size_t)p * CIN_ + sub * 16;
    float* sdst = sums + (size_t)lin * CIN_ + sub * 16;
#pragma unroll
    for (int c = 0; c < 16; ++c) atomicAdd(&sdst[c], f[c]);
}

// ---------- 3: grid average -> bf16 ----------
__global__ void k_finalize(const float* __restrict__ sums, const float* __restrict__ cnts,
                           unsigned short* __restrict__ grid0) {
    unsigned int t = blockIdx.x * blockDim.x + threadIdx.x;
    if (t >= NVOX * 8) return;
    unsigned int part = t & 7u;
    unsigned int vox = t >> 3;
    float cnt = cnts[vox];
    float inv = (cnt == 0.f) ? 1.f : 1.f / cnt;
    const float* s = sums + (size_t)vox * CIN_ + part * 8;
    unsigned short* d = grid0 + (size_t)vox * CIN_ + part * 8;
#pragma unroll
    for (int c = 0; c < 8; ++c) d[c] = f2bf(s[c] * inv);
}

// ---------- 4: pack conv weights into per-lane WMMA B fragments ----------
// GEMM B[k][n], k = neighbor*CIN + cin (matches (K,K,K,CIN,COUT) row-major), n = cout.
// Fragment record (kstep s, ntile nt): 32 lanes x 8 dwords; lane: n = nt*16 + (l&15),
// dword v holds bf16 pair for k = s*32 + (l>>4)*16 + 2v (+1 in high half).
__global__ void k_pack_w(const float* __restrict__ w, unsigned int* __restrict__ wp, int ksteps) {
    unsigned int idx = blockIdx.x * blockDim.x + threadIdx.x;
    if (idx >= (unsigned int)(ksteps * 8 * 32 * 8)) return;
    unsigned int v    = idx & 7u;
    unsigned int lane = (idx >> 3) & 31u;
    unsigned int nt   = (idx >> 8) & 7u;
    unsigned int s    = idx >> 11;
    unsigned int n = nt * 16 + (lane & 15u);
    unsigned int k = s * 32 + (lane >> 4) * 16 + 2 * v;
    unsigned short lo = f2bf(w[(size_t)k * COUT_ + n]);
    unsigned short hh = f2bf(w[(size_t)(k + 1) * COUT_ + n]);
    wp[idx] = (unsigned int)lo | ((unsigned int)hh << 16);
}

// ---------- 5: implicit-GEMM 3x3x3 conv via v_wmma_f32_16x16x32_bf16 ----------
// Wave tile: 32 voxels (full W row) x 128 channels: two A fragments share each B fragment.
// Epilogue: +bias, BN(eps 1e-4), LeakyReLU(0.1), bf16 out.
template <int CS>   // cin steps of 32 (2 for cin=64, 4 for cin=128)
__global__ __launch_bounds__(256) void k_conv3d(
    const unsigned int* __restrict__ in_dw,   // bf16 grid viewed as dword pairs, [voxel][cin/2]
    const uint4* __restrict__ wp,             // packed fragments
    unsigned short* __restrict__ out_bf,      // [voxel][128] bf16
    const float* __restrict__ bias, const float* __restrict__ bn_g,
    const float* __restrict__ bn_b, const float* __restrict__ bn_m,
    const float* __restrict__ bn_v) {

    unsigned int wave = (blockIdx.x << 3) + (threadIdx.x >> 5);   // 0..4095
    unsigned int lane = threadIdx.x & 31u;
    unsigned int mrow = lane & 15u;
    unsigned int hi   = lane >> 4;
    unsigned int h = wave & 31u;
    unsigned int d = (wave >> 5) & 31u;
    unsigned int b = wave >> 10;
    unsigned int vox_base = ((b * R_ + d) * R_ + h) * R_;   // 32 voxels along W

    float sc[8], sh[8];
#pragma unroll
    for (int nt = 0; nt < 8; ++nt) {
        int c = nt * 16 + (int)mrow;
        float s = bn_g[c] * rsqrtf(bn_v[c] + 1e-4f);
        sc[nt] = s;
        sh[nt] = (bias[c] - bn_m[c]) * s + bn_b[c];
    }

    v8f acc0[8], acc1[8];
    v8f zero = {0.f, 0.f, 0.f, 0.f, 0.f, 0.f, 0.f, 0.f};
#pragma unroll
    for (int i = 0; i < 8; ++i) { acc0[i] = zero; acc1[i] = zero; }

    for (int nb = 0; nb < 27; ++nb) {
        int zd = (int)d + nb / 9 - 1;
        int zh = (int)h + (nb / 3) % 3 - 1;
        int dw = nb % 3 - 1;
        int zw0 = (int)mrow + dw;            // A-row voxel W for half 0
        int zw1 = zw0 + 16;                  // half 1
        bool vdh = ((unsigned)zd < (unsigned)R_) & ((unsigned)zh < (unsigned)R_);
        bool valid0 = vdh & ((unsigned)zw0 < (unsigned)R_);
        bool valid1 = vdh & ((unsigned)zw1 < (unsigned)R_);
        unsigned int row_base = ((b * R_ + (unsigned)(zd & 31)) * R_ + (unsigned)(zh & 31)) * R_;
        unsigned int nvox0 = row_base + (unsigned)(zw0 & 31);
        unsigned int nvox1 = row_base + (unsigned)(zw1 & 31);
        const unsigned int* abase0 = in_dw + (size_t)nvox0 * (CS << 4);
        const unsigned int* abase1 = in_dw + (size_t)nvox1 * (CS << 4);
#pragma unroll 1
        for (int cs = 0; cs < CS; ++cs) {
            int sstep = nb * CS + cs;
            union { unsigned int u[8]; vbf16 v; } A0, A1;
#pragma unroll
            for (int vr = 0; vr < 8; ++vr) {
                // A kpair layout: K = vr*2 + hi*16
                unsigned int k0 = ((unsigned)vr << 1) + (hi << 4);
                unsigned int off = ((unsigned)cs << 4) + (k0 >> 1);
                unsigned int t0 = abase0[off];        // address always in-range (clamped)
                unsigned int t1 = abase1[off];
                A0.u[vr] = valid0 ? t0 : 0u;          // branchless zero-fill
                A1.u[vr] = valid1 ? t1 : 0u;
            }
            const uint4* wpt = wp + ((size_t)sstep << 9) + (lane << 1);
#pragma unroll
            for (int nt = 0; nt < 8; ++nt) {
                union { uint4 q[2]; vbf16 v; } Bf;
                Bf.q[0] = wpt[nt << 6];
                Bf.q[1] = wpt[(nt << 6) + 1];
                acc0[nt] = __builtin_amdgcn_wmma_f32_16x16x32_bf16(
                    false, A0.v, false, Bf.v, (short)0, acc0[nt], false, false);
                acc1[nt] = __builtin_amdgcn_wmma_f32_16x16x32_bf16(
                    false, A1.v, false, Bf.v, (short)0, acc1[nt], false, false);
            }
        }
    }

    // Epilogue: D layout — lane column n = lane&15, row m = j + 8*(lane>>4)
#pragma unroll
    for (int nt = 0; nt < 8; ++nt) {
        unsigned int c = nt * 16 + mrow;
#pragma unroll
        for (int j = 0; j < 8; ++j) {
            unsigned int m = (hi << 3) + j;
            float x0 = acc0[nt][j] * sc[nt] + sh[nt];
            x0 = (x0 >= 0.f) ? x0 : 0.1f * x0;
            out_bf[(size_t)(vox_base + m) * COUT_ + c] = f2bf(x0);
            float x1 = acc1[nt][j] * sc[nt] + sh[nt];
            x1 = (x1 >= 0.f) ? x1 : 0.1f * x1;
            out_bf[(size_t)(vox_base + 16 + m) * COUT_ + c] = f2bf(x1);
        }
    }
}

// ---------- 6: trilinear devoxelize + point MLP + add ----------
__global__ void k_devox_point(const unsigned short* __restrict__ g2,
                              const float* __restrict__ nc,
                              const float* __restrict__ feat,
                              const float* __restrict__ pw, const float* __restrict__ pb,
                              const float* __restrict__ bn_g, const float* __restrict__ bn_b,
                              const float* __restrict__ bn_m, const float* __restrict__ bn_v,
                              float* __restrict__ out) {
    unsigned int t = blockIdx.x * blockDim.x + threadIdx.x;
    if (t >= B_ * N_ * 8) return;
    unsigned int cg = t & 7u;           // 16-channel group
    unsigned int p  = t >> 3;
    unsigned int b  = p >> 14;

    float nx = nc[p * 3 + 0], ny = nc[p * 3 + 1], nz = nc[p * 3 + 2];
    int x0 = (int)floorf(nx), y0 = (int)floorf(ny), z0 = (int)floorf(nz);
    float fx = nx - (float)x0, fy = ny - (float)y0, fz = nz - (float)z0;
    int x1 = min(x0 + 1, R_ - 1), y1 = min(y0 + 1, R_ - 1), z1 = min(z0 + 1, R_ - 1);

    float acc[16];
#pragma unroll
    for (int c = 0; c < 16; ++c) acc[c] = 0.f;

    for (int corner = 0; corner < 8; ++corner) {
        int ix = (corner & 4) ? x1 : x0;
        int iy = (corner & 2) ? y1 : y0;
        int iz = (corner & 1) ? z1 : z0;
        float wgt = ((corner & 4) ? fx : 1.f - fx) *
                    ((corner & 2) ? fy : 1.f - fy) *
                    ((corner & 1) ? fz : 1.f - fz);
        const unsigned short* gp =
            g2 + ((size_t)(b * (R_ * R_ * R_) + ix * (R_ * R_) + iy * R_ + iz)) * COUT_ + cg * 16;
#pragma unroll
        for (int c = 0; c < 16; ++c) acc[c] += wgt * bf2f(gp[c]);
    }

    // point branch: features @ pw (fp32)
    const float* fp = feat + (size_t)p * CIN_;
    float pf[16];
#pragma unroll
    for (int c = 0; c < 16; ++c) pf[c] = 0.f;
    for (int ci = 0; ci < CIN_; ++ci) {
        float f = fp[ci];
        const float* wrow = pw + (size_t)ci * COUT_ + cg * 16;
#pragma unroll
        for (int c = 0; c < 16; ++c) pf[c] += f * wrow[c];
    }
#pragma unroll
    for (int c = 0; c < 16; ++c) {
        unsigned int ch = cg * 16 + c;
        float s = bn_g[ch] * rsqrtf(bn_v[ch] + 1e-5f);
        float x = (pf[c] + pb[ch] - bn_m[ch]) * s + bn_b[ch];
        x = fmaxf(x, 0.f);
        out[(size_t)p * COUT_ + ch] = acc[c] + x;
    }
}

// ---------- host launch ----------
extern "C" void kernel_launch(void* const* d_in, const int* in_sizes, int n_in,
                              void* d_out, int out_size, void* d_ws, size_t ws_size,
                              hipStream_t stream) {
    const float* features = (const float*)d_in[0];
    const float* coords   = (const float*)d_in[1];
    const float* conv1_w  = (const float*)d_in[2];
    const float* conv1_b  = (const float*)d_in[3];
    const float* bn1_g = (const float*)d_in[4];
    const float* bn1_b = (const float*)d_in[5];
    const float* bn1_m = (const float*)d_in[6];
    const float* bn1_v = (const float*)d_in[7];
    const float* conv2_w  = (const float*)d_in[8];
    const float* conv2_b  = (const float*)d_in[9];
    const float* bn2_g = (const float*)d_in[10];
    const float* bn2_b = (const float*)d_in[11];
    const float* bn2_m = (const float*)d_in[12];
    const float* bn2_v = (const float*)d_in[13];
    const float* pw    = (const float*)d_in[14];
    const float* pb    = (const float*)d_in[15];
    const float* pbn_g = (const float*)d_in[16];
    const float* pbn_b = (const float*)d_in[17];
    const float* pbn_m = (const float*)d_in[18];
    const float* pbn_v = (const float*)d_in[19];
    float* out = (float*)d_out;
    char* ws = (char*)d_ws;

    // workspace layout (256B-aligned)
    const size_t o_stats = 0;                               //  64 B (B*4 f32)
    const size_t o_nc    = 256;                             //  786,432 B
    const size_t o_w1p   = o_nc   + 786432;                 //  442,368 B
    const size_t o_w2p   = o_w1p  + 442368;                 //  884,736 B
    const size_t o_cnts  = o_w2p  + 884736;                 //  524,288 B
    const size_t o_sums  = o_cnts + 524288;                 //  33,554,432 B (reused as g2 bf16)
    const size_t o_grid0 = o_sums + 33554432;               //  16,777,216 B
    const size_t o_g1    = o_grid0 + 16777216;              //  33,554,432 B

    float*          stats = (float*)(ws + o_stats);
    float*          ncbuf = (float*)(ws + o_nc);
    unsigned int*   w1p   = (unsigned int*)(ws + o_w1p);
    unsigned int*   w2p   = (unsigned int*)(ws + o_w2p);
    float*          cnts  = (float*)(ws + o_cnts);
    float*          sums  = (float*)(ws + o_sums);
    unsigned short* grid0 = (unsigned short*)(ws + o_grid0);
    unsigned short* g1    = (unsigned short*)(ws + o_g1);
    unsigned short* g2    = (unsigned short*)(ws + o_sums);   // reuse sums region

    // 0) zero cnts + sums (contiguous region)
    size_t zero_bytes = 524288 + 33554432;
    k_zero<<<2048, 256, 0, stream>>>((uint4*)(ws + o_cnts), zero_bytes / 16);

    // 1) per-batch coord stats
    k_stats<<<B_, 256, 0, stream>>>(coords, stats);

    // 2) voxelize scatter
    k_voxelize<<<(B_ * N_ * 4) / 256, 256, 0, stream>>>(coords, features, stats, ncbuf, sums, cnts);

    // 3) average -> bf16 grid
    k_finalize<<<(NVOX * 8) / 256, 256, 0, stream>>>(sums, cnts, grid0);

    // 4) pack weights (conv1: 54 ksteps, conv2: 108 ksteps)
    k_pack_w<<<(54 * 2048) / 256, 256, 0, stream>>>(conv1_w, w1p, 54);
    k_pack_w<<<(108 * 2048) / 256, 256, 0, stream>>>(conv2_w, w2p, 108);

    // 5) conv1 (cin=64 -> 2 cin steps), conv2 (cin=128 -> 4 cin steps)
    //    4096 wave tasks (32 voxels x 128 ch each) -> 512 blocks of 8 waves
    k_conv3d<2><<<512, 256, 0, stream>>>((const unsigned int*)grid0, (const uint4*)w1p, g1,
                                         conv1_b, bn1_g, bn1_b, bn1_m, bn1_v);
    k_conv3d<4><<<512, 256, 0, stream>>>((const unsigned int*)g1, (const uint4*)w2p, g2,
                                         conv2_b, bn2_g, bn2_b, bn2_m, bn2_v);

    // 6) devoxelize + point MLP + add
    k_devox_point<<<(B_ * N_ * 8) / 256, 256, 0, stream>>>(g2, ncbuf, features, pw, pb,
                                                           pbn_g, pbn_b, pbn_m, pbn_v, out);

    // 7) coords passthrough (second tuple element)
    hipMemcpyAsync(out + (size_t)B_ * N_ * COUT_, coords,
                   (size_t)B_ * N_ * 3 * sizeof(float), hipMemcpyDeviceToDevice, stream);
}
